// GNNVariationalAD_42090679501241
// MI455X (gfx1250) — matrix operand order repaired and checked by reading, not compiled, not run on hardware
//
#include <hip/hip_runtime.h>
#include <hip/hip_bf16.h>

// ---------------------------------------------------------------------------
// GNN-VAE anomaly detector for MI455X (gfx1250, wave32, WMMA f32_16x16x32_f16)
// B=64, T=256, C=64, GNN=32, H=64, L=32
// ---------------------------------------------------------------------------

typedef __attribute__((ext_vector_type(16))) _Float16 v16h;
typedef __attribute__((ext_vector_type(8)))  float    v8f;
typedef __attribute__((ext_vector_type(4)))  int      v4i;

#define B_   64
#define T_   256
#define C_   64
#define GNN_ 32
#define H_   64
#define L_   32

struct alignas(16) Q16 { unsigned long long a, b; };

#if defined(__has_builtin)
#if __has_builtin(__builtin_amdgcn_global_load_async_to_lds_b128) && \
    __has_builtin(__builtin_amdgcn_s_wait_asynccnt)
#define HAVE_ASYNC_LDS 1
#endif
#endif

#ifdef HAVE_ASYNC_LDS
typedef __attribute__((address_space(1))) void gas_void;
typedef __attribute__((address_space(3))) void las_void;
typedef __attribute__((address_space(1))) v4i* gas_v4i_ptr;
typedef __attribute__((address_space(3))) v4i* las_v4i_ptr;
#endif

// Stage nhalf f16 elements from global to LDS (16B granules), cooperative.
// Uses CDNA5 GLOBAL_LOAD_ASYNC_TO_LDS_B128 (ASYNCcnt) when available.
__device__ __forceinline__ void stage_to_lds(_Float16* dst,
                                             const _Float16* src, int nhalf,
                                             int tid, int nthreads) {
#ifdef HAVE_ASYNC_LDS
  for (int i = tid * 8; i < nhalf; i += nthreads * 8) {
    void* sp = const_cast<_Float16*>(src) + i;
    gas_void* gs = (gas_void*)sp;          // addrspacecast AS0 -> AS1
    las_void* ls = (las_void*)(void*)(dst + i);  // addrspacecast AS0 -> AS3
    __builtin_amdgcn_global_load_async_to_lds_b128(
        (gas_v4i_ptr)gs, (las_v4i_ptr)ls, 0, 0);
  }
  __builtin_amdgcn_s_wait_asynccnt(0);
#else
  for (int i = tid * 8; i < nhalf; i += nthreads * 8)
    *(Q16*)(dst + i) = *(const Q16*)(src + i);
#endif
}

// A-fragment (16x32, f16): lane<16 -> M=lane, K in {kb..kb+7, kb+16..kb+23};
// lane>=16 -> same M, K shifted by +8 within each half (per CDNA5 ISA 7.12.2).
__device__ __forceinline__ v16h load_a_frag(const _Float16* base, int ld,
                                            int kbase, int lane) {
  const int m  = lane & 15;
  const int hi = (lane >> 4) & 1;
  const _Float16* rp = base + (size_t)m * ld + kbase + hi * 8;
  union { v16h v; Q16 q[2]; } f;
  f.q[0] = *(const Q16*)(rp);
  f.q[1] = *(const Q16*)(rp + 16);
  return f.v;
}

// B-fragment loaded from B^T stored row-major (N x K). lane<16 -> N=lane,
// K in {kb..kb+7, kb+16..kb+23}; lane>=16 -> same N, K shifted by +8 (matches A).
__device__ __forceinline__ v16h load_bt_frag(const _Float16* base, int ld,
                                             int kbase, int lane) {
  const int n  = lane & 15;
  const int hi = (lane >> 4) & 1;
  const _Float16* rp = base + (size_t)n * ld + kbase + hi * 8;
  union { v16h v; Q16 q[2]; } f;
  f.q[0] = *(const Q16*)(rp);
  f.q[1] = *(const Q16*)(rp + 16);
  return f.v;
}

__device__ __forceinline__ v8f wmma16(v16h a, v16h b, v8f c) {
  return __builtin_amdgcn_wmma_f32_16x16x32_f16(false, a, false, b,
                                                (short)0, c, false, false);
}

__device__ __forceinline__ float sigf(float x) {
  return 1.f / (1.f + __expf(-x));
}
__device__ __forceinline__ float tanhf_(float x) {
  float e = __expf(2.f * x);
  return (e - 1.f) / (e + 1.f);
}

// ---------------------------------------------------------------------------
// Prep: row softmax of 64x64 adjacency -> f16; f32 -> f16 weight conversion
// ---------------------------------------------------------------------------
__global__ void __launch_bounds__(64)
softmax64_kernel(const float* __restrict__ a, _Float16* __restrict__ o) {
  const int r = threadIdx.x;
  float mx = -1e30f;
  for (int j = 0; j < 64; j++) mx = fmaxf(mx, a[r * 64 + j]);
  float s = 0.f;
  for (int j = 0; j < 64; j++) s += __expf(a[r * 64 + j] - mx);
  float inv = 1.f / s;
  for (int j = 0; j < 64; j++)
    o[r * 64 + j] = (_Float16)(__expf(a[r * 64 + j] - mx) * inv);
}

__global__ void __launch_bounds__(256)
cvt_f16_kernel(const float* __restrict__ s, _Float16* __restrict__ d, int n) {
  int i = blockIdx.x * 256 + threadIdx.x;
  if (i < n) d[i] = (_Float16)s[i];
}

// ---------------------------------------------------------------------------
// GNN encoder: per (b,t) item: s = A1s@x ; h1 = relu(s*w1+b1);
// T1 = h1 @ W2^T (WMMA); h2 = relu(A2s @ T1 + b2) (WMMA); LayerNorm;
// store h_enc[(b*64+c), t, g] f16.   128 threads = 4 waves, 1 item / block.
// ---------------------------------------------------------------------------
__global__ void __launch_bounds__(128)
gnn_kernel(const float* __restrict__ x, const _Float16* __restrict__ A1s,
           const _Float16* __restrict__ A2s, const _Float16* __restrict__ W2h,
           const float* __restrict__ W1, const float* __restrict__ b1,
           const float* __restrict__ b2, const float* __restrict__ ln_g,
           const float* __restrict__ ln_b, _Float16* __restrict__ h_enc) {
  __shared__ float xs[64];
  __shared__ float sv[64];
  __shared__ alignas(16) _Float16 h1[64 * 32];
  __shared__ alignas(16) _Float16 T1t[32 * 64];   // T1 transposed (N x M)
  __shared__ float h2[64 * 32];

  const int item = blockIdx.x;
  const int b = item >> 8;
  const int t = item & 255;
  const int tid = threadIdx.x;
  const int lane = tid & 31, wv = tid >> 5;
  const int hi = (lane >> 4) & 1, l15 = lane & 15;

  if (tid < 64) xs[tid] = x[((size_t)b * T_ + t) * C_ + tid];
  __syncthreads();

  if (tid < 64) {
    float acc = 0.f;
    #pragma unroll 8
    for (int j = 0; j < 64; j++) acc += (float)A1s[tid * 64 + j] * xs[j];
    sv[tid] = acc;
  }
  __syncthreads();

  #pragma unroll
  for (int e = 0; e < 16; e++) {
    int idx = tid * 16 + e;
    int i = idx >> 5, g = idx & 31;
    h1[idx] = (_Float16)fmaxf(sv[i] * W1[g] + b1[g], 0.f);
  }
  __syncthreads();

  // T1 = h1(64x32) @ W2^T(32x32): wave wv -> M-tile wv, N-tiles 0..1
  #pragma unroll
  for (int nt = 0; nt < 2; nt++) {
    v16h a = load_a_frag(&h1[(wv * 16) * 32], 32, 0, lane);
    v16h bf = load_bt_frag(&W2h[(nt * 16) * 32], 32, 0, lane);
    v8f c;
    #pragma unroll
    for (int e = 0; e < 8; e++) c[e] = 0.f;
    c = wmma16(a, bf, c);
    #pragma unroll
    for (int e = 0; e < 8; e++) {
      int m = wv * 16 + e + hi * 8;
      int n = nt * 16 + l15;
      T1t[n * 64 + m] = (_Float16)c[e];
    }
  }
  __syncthreads();

  // h2 = A2s(64x64) @ T1(64x32)
  #pragma unroll
  for (int nt = 0; nt < 2; nt++) {
    v8f c;
    #pragma unroll
    for (int e = 0; e < 8; e++) c[e] = 0.f;
    #pragma unroll
    for (int kk = 0; kk < 2; kk++) {
      v16h a = load_a_frag(&A2s[(size_t)(wv * 16) * 64], 64, kk * 32, lane);
      v16h bf = load_bt_frag(&T1t[(nt * 16) * 64], 64, kk * 32, lane);
      c = wmma16(a, bf, c);
    }
    #pragma unroll
    for (int e = 0; e < 8; e++) {
      int m = wv * 16 + e + hi * 8;
      int g = nt * 16 + l15;
      h2[m * 32 + g] = fmaxf(c[e] + b2[g], 0.f);
    }
  }
  __syncthreads();

  if (tid < 64) {
    float mean = 0.f;
    for (int g = 0; g < 32; g++) mean += h2[tid * 32 + g];
    mean *= (1.f / 32.f);
    float var = 0.f;
    for (int g = 0; g < 32; g++) {
      float d = h2[tid * 32 + g] - mean;
      var += d * d;
    }
    var *= (1.f / 32.f);
    float inv = rsqrtf(var + 1e-5f);
    size_t node = (size_t)b * 64 + tid;
    _Float16* dst = h_enc + (node * T_ + t) * GNN_;
    for (int g = 0; g < 32; g++)
      dst[g] = (_Float16)((h2[tid * 32 + g] - mean) * inv * ln_g[g] + ln_b[g]);
  }
}

// ---------------------------------------------------------------------------
// GRU direction scan. 4 waves/block, each wave owns 16 rows (nodes) and scans
// all timesteps with its private double-buffered LDS h-tile. Gates r,z,n are
// N-tiles {ct, 4+ct, 8+ct} of the 192-wide gate matrix. Weights are staged
// once into (dynamic) LDS via async global->LDS copy and all B-fragments are
// served from LDS for the whole scan.
// ---------------------------------------------------------------------------
template <int KIN, bool REV>
__global__ void __launch_bounds__(128)
gru_kernel(const _Float16* __restrict__ x,    // [4096][256][KIN]
           const _Float16* __restrict__ Wih,  // [192][KIN] f16
           const _Float16* __restrict__ Whh,  // [192][64]  f16
           const float* __restrict__ bih, const float* __restrict__ bhh,
           int steps,
           _Float16* __restrict__ yseq, int ycol,      // [4096][256][128]
           _Float16* __restrict__ ylast, int ycol_last) // [4096][128]
{
  __shared__ alignas(16) _Float16 hbuf[2][4][16 * 64];
  extern __shared__ __align__(16) _Float16 smem[];
  _Float16* sWih = smem;                 // [192][KIN]
  _Float16* sWhh = smem + 192 * KIN;     // [192][64]

  const int tid = threadIdx.x, lane = tid & 31, wv = tid >> 5;
  const int hi = (lane >> 4) & 1, l15 = lane & 15;
  const int node0 = (blockIdx.x * 4 + wv) * 16;

  stage_to_lds(sWih, Wih, 192 * KIN, tid, 128);
  stage_to_lds(sWhh, Whh, 192 * 64, tid, 128);
  {
    _Float16* p = &hbuf[0][0][0];
    for (int i = tid; i < 4 * 16 * 64; i += 128) p[i] = (_Float16)0.f;
  }
  float bi[12], bh[12];
  #pragma unroll
  for (int j = 0; j < 12; j++) {
    bi[j] = bih[j * 16 + l15];
    bh[j] = bhh[j * 16 + l15];
  }
  v8f hreg[4];
  #pragma unroll
  for (int c = 0; c < 4; c++)
    #pragma unroll
    for (int e = 0; e < 8; e++) hreg[c][e] = 0.f;
  __syncthreads();

  const size_t xrow = (size_t)T_ * KIN;
  for (int s = 0; s < steps; s++) {
    const int t = REV ? (T_ - 1 - s) : s;
    const int cur = s & 1, nxt = cur ^ 1;
    const _Float16* xb = x + (size_t)node0 * xrow + (size_t)t * KIN;

    v16h xa[KIN / 32];
    #pragma unroll
    for (int kk = 0; kk < KIN / 32; kk++)
      xa[kk] = load_a_frag(xb, (int)xrow, kk * 32, lane);
    v16h ha0 = load_a_frag(&hbuf[cur][wv][0], 64, 0, lane);
    v16h ha1 = load_a_frag(&hbuf[cur][wv][0], 64, 32, lane);

    if (s + 1 < steps) {
      const int tn = REV ? (T_ - 2 - s) : (s + 1);
      __builtin_prefetch(x + (size_t)(node0 + l15) * xrow + (size_t)tn * KIN, 0, 3);
    }

    #pragma unroll
    for (int ct = 0; ct < 4; ct++) {
      v8f xr, hr, xz, hz, xn, hn;
      #pragma unroll
      for (int e = 0; e < 8; e++) {
        xr[e] = bi[ct];     hr[e] = bh[ct];
        xz[e] = bi[4 + ct]; hz[e] = bh[4 + ct];
        xn[e] = bi[8 + ct]; hn[e] = bh[8 + ct];
      }
      #pragma unroll
      for (int kk = 0; kk < KIN / 32; kk++) {
        xr = wmma16(xa[kk], load_bt_frag(&sWih[(size_t)(ct * 16) * KIN], KIN, kk * 32, lane), xr);
        xz = wmma16(xa[kk], load_bt_frag(&sWih[(size_t)((4 + ct) * 16) * KIN], KIN, kk * 32, lane), xz);
        xn = wmma16(xa[kk], load_bt_frag(&sWih[(size_t)((8 + ct) * 16) * KIN], KIN, kk * 32, lane), xn);
      }
      hr = wmma16(ha0, load_bt_frag(&sWhh[(ct * 16) * 64], 64, 0, lane), hr);
      hr = wmma16(ha1, load_bt_frag(&sWhh[(ct * 16) * 64], 64, 32, lane), hr);
      hz = wmma16(ha0, load_bt_frag(&sWhh[((4 + ct) * 16) * 64], 64, 0, lane), hz);
      hz = wmma16(ha1, load_bt_frag(&sWhh[((4 + ct) * 16) * 64], 64, 32, lane), hz);
      hn = wmma16(ha0, load_bt_frag(&sWhh[((8 + ct) * 16) * 64], 64, 0, lane), hn);
      hn = wmma16(ha1, load_bt_frag(&sWhh[((8 + ct) * 16) * 64], 64, 32, lane), hn);

      #pragma unroll
      for (int e = 0; e < 8; e++) {
        float r = sigf(xr[e] + hr[e]);
        float z = sigf(xz[e] + hz[e]);
        float n = tanhf_(xn[e] + r * hn[e]);
        float hnew = (1.f - z) * n + z * hreg[ct][e];
        hreg[ct][e] = hnew;
        int row = e + hi * 8, col = ct * 16 + l15;
        hbuf[nxt][wv][row * 64 + col] = (_Float16)hnew;
        if (yseq)
          yseq[((size_t)(node0 + row) * T_ + t) * 128 + ycol + col] = (_Float16)hnew;
      }
    }
    __syncthreads();
  }

  if (ylast) {
    #pragma unroll
    for (int ct = 0; ct < 4; ct++)
      #pragma unroll
      for (int e = 0; e < 8; e++) {
        int row = e + hi * 8, col = ct * 16 + l15;
        ylast[(size_t)(node0 + row) * 128 + ycol_last + col] = (_Float16)hreg[ct][e];
      }
  }
}

// ---------------------------------------------------------------------------
// Pool + VAE head + decoder fc. Writes mu/logv to d_out, dh (f16) for LSTM.
// ---------------------------------------------------------------------------
__global__ void __launch_bounds__(256)
pool_kernel(const _Float16* __restrict__ hlast,  // [4096][128]
            const float* __restrict__ pool_W, const float* __restrict__ pool_b,
            const float* __restrict__ mu_W, const float* __restrict__ mu_b,
            const float* __restrict__ logv_W, const float* __restrict__ logv_b,
            const float* __restrict__ dec_W, const float* __restrict__ dec_b,
            float* __restrict__ out_mu, float* __restrict__ out_logv,
            _Float16* __restrict__ dh) {
  __shared__ float lastS[64 * 128];
  __shared__ float hpS[64 * 32];
  __shared__ float muS[64 * 32];
  const int tid = threadIdx.x;

  for (int idx = tid; idx < 64 * 128; idx += 256) {
    int b = idx >> 7, k = idx & 127;
    float acc = 0.f;
    for (int c = 0; c < 64; c++)
      acc += (float)hlast[(size_t)(b * 64 + c) * 128 + k];
    lastS[idx] = acc * (1.f / 64.f);
  }
  __syncthreads();
  for (int idx = tid; idx < 64 * 32; idx += 256) {
    int b = idx >> 5, l = idx & 31;
    float acc = pool_b[l];
    for (int k = 0; k < 128; k++) acc += lastS[b * 128 + k] * pool_W[l * 128 + k];
    hpS[idx] = fmaxf(acc, 0.f);
  }
  __syncthreads();
  for (int idx = tid; idx < 64 * 32; idx += 256) {
    int b = idx >> 5, l2 = idx & 31;
    float am = mu_b[l2], av = logv_b[l2];
    for (int l = 0; l < 32; l++) {
      float h = hpS[b * 32 + l];
      am += h * mu_W[l2 * 32 + l];
      av += h * logv_W[l2 * 32 + l];
    }
    muS[idx] = am;
    out_mu[idx] = am;
    out_logv[idx] = av;
  }
  __syncthreads();
  for (int idx = tid; idx < 64 * 64; idx += 256) {
    int b = idx >> 6, hh = idx & 63;
    float acc = dec_b[hh];
    for (int l = 0; l < 32; l++) acc += muS[b * 32 + l] * dec_W[hh * 32 + l];
    dh[idx] = (_Float16)fmaxf(acc, 0.f);
  }
}

// ---------------------------------------------------------------------------
// LSTM layer 0: input constant over time (repeated latent) -> precompute xg.
// 1 block, 4 waves x 16 rows (M=64). Gates i,f,g,o = tiles {ct,4+ct,8+ct,12+ct}.
// Weights staged in dynamic LDS (async copy).
// ---------------------------------------------------------------------------
__global__ void __launch_bounds__(128)
lstm_first_kernel(const _Float16* __restrict__ dh,   // [64][64]
                  const _Float16* __restrict__ Wih,  // [256][64]
                  const _Float16* __restrict__ Whh,  // [256][64]
                  const float* __restrict__ bih, const float* __restrict__ bhh,
                  _Float16* __restrict__ yseq)       // [64][256][64]
{
  __shared__ alignas(16) _Float16 hbuf[2][4][16 * 64];
  extern __shared__ __align__(16) _Float16 smem[];
  _Float16* sWih = smem;              // [256][64]
  _Float16* sWhh = smem + 256 * 64;   // [256][64]
  const int tid = threadIdx.x, lane = tid & 31, wv = tid >> 5;
  const int hi = (lane >> 4) & 1, l15 = lane & 15;
  const int row0 = wv * 16;
  stage_to_lds(sWih, Wih, 256 * 64, tid, 128);
  stage_to_lds(sWhh, Whh, 256 * 64, tid, 128);
  {
    _Float16* p = &hbuf[0][0][0];
    for (int i = tid; i < 4 * 16 * 64; i += 128) p[i] = (_Float16)0.f;
  }
  float bs[16];
  #pragma unroll
  for (int j = 0; j < 16; j++) bs[j] = bih[j * 16 + l15] + bhh[j * 16 + l15];
  __syncthreads();

  v16h a0 = load_a_frag(&dh[(size_t)row0 * 64], 64, 0, lane);
  v16h a1 = load_a_frag(&dh[(size_t)row0 * 64], 64, 32, lane);
  v8f xg[16];
  #pragma unroll
  for (int j = 0; j < 16; j++) {
    v8f c;
    #pragma unroll
    for (int e = 0; e < 8; e++) c[e] = bs[j];
    c = wmma16(a0, load_bt_frag(&sWih[(j * 16) * 64], 64, 0, lane), c);
    c = wmma16(a1, load_bt_frag(&sWih[(j * 16) * 64], 64, 32, lane), c);
    xg[j] = c;
  }
  v8f creg[4];
  #pragma unroll
  for (int c = 0; c < 4; c++)
    #pragma unroll
    for (int e = 0; e < 8; e++) creg[c][e] = 0.f;

  for (int t = 0; t < T_; t++) {
    const int cur = t & 1, nxt = cur ^ 1;
    v16h h0 = load_a_frag(&hbuf[cur][wv][0], 64, 0, lane);
    v16h h1 = load_a_frag(&hbuf[cur][wv][0], 64, 32, lane);
    #pragma unroll
    for (int ct = 0; ct < 4; ct++) {
      v8f gi = xg[ct], gf = xg[4 + ct], gg = xg[8 + ct], go = xg[12 + ct];
      gi = wmma16(h0, load_bt_frag(&sWhh[(ct * 16) * 64], 64, 0, lane), gi);
      gi = wmma16(h1, load_bt_frag(&sWhh[(ct * 16) * 64], 64, 32, lane), gi);
      gf = wmma16(h0, load_bt_frag(&sWhh[((4 + ct) * 16) * 64], 64, 0, lane), gf);
      gf = wmma16(h1, load_bt_frag(&sWhh[((4 + ct) * 16) * 64], 64, 32, lane), gf);
      gg = wmma16(h0, load_bt_frag(&sWhh[((8 + ct) * 16) * 64], 64, 0, lane), gg);
      gg = wmma16(h1, load_bt_frag(&sWhh[((8 + ct) * 16) * 64], 64, 32, lane), gg);
      go = wmma16(h0, load_bt_frag(&sWhh[((12 + ct) * 16) * 64], 64, 0, lane), go);
      go = wmma16(h1, load_bt_frag(&sWhh[((12 + ct) * 16) * 64], 64, 32, lane), go);
      #pragma unroll
      for (int e = 0; e < 8; e++) {
        float ii = sigf(gi[e]), ff = sigf(gf[e]);
        float g_ = tanhf_(gg[e]), oo = sigf(go[e]);
        float cc = ff * creg[ct][e] + ii * g_;
        creg[ct][e] = cc;
        float hh = oo * tanhf_(cc);
        int row = e + hi * 8, col = ct * 16 + l15;
        hbuf[nxt][wv][row * 64 + col] = (_Float16)hh;
        yseq[((size_t)(row0 + row) * T_ + t) * 64 + col] = (_Float16)hh;
      }
    }
    __syncthreads();
  }
}

// LSTM layer 1: input varies per t (reads layer-0 output sequence).
__global__ void __launch_bounds__(128)
lstm_seq_kernel(const _Float16* __restrict__ xseq,  // [64][256][64]
                const _Float16* __restrict__ Wih, const _Float16* __restrict__ Whh,
                const float* __restrict__ bih, const float* __restrict__ bhh,
                _Float16* __restrict__ yseq) {
  __shared__ alignas(16) _Float16 hbuf[2][4][16 * 64];
  extern __shared__ __align__(16) _Float16 smem[];
  _Float16* sWih = smem;              // [256][64]
  _Float16* sWhh = smem + 256 * 64;   // [256][64]
  const int tid = threadIdx.x, lane = tid & 31, wv = tid >> 5;
  const int hi = (lane >> 4) & 1, l15 = lane & 15;
  const int row0 = wv * 16;
  stage_to_lds(sWih, Wih, 256 * 64, tid, 128);
  stage_to_lds(sWhh, Whh, 256 * 64, tid, 128);
  {
    _Float16* p = &hbuf[0][0][0];
    for (int i = tid; i < 4 * 16 * 64; i += 128) p[i] = (_Float16)0.f;
  }
  float bs[16];
  #pragma unroll
  for (int j = 0; j < 16; j++) bs[j] = bih[j * 16 + l15] + bhh[j * 16 + l15];
  v8f creg[4];
  #pragma unroll
  for (int c = 0; c < 4; c++)
    #pragma unroll
    for (int e = 0; e < 8; e++) creg[c][e] = 0.f;
  __syncthreads();

  const size_t xrow = (size_t)T_ * 64;
  for (int t = 0; t < T_; t++) {
    const int cur = t & 1, nxt = cur ^ 1;
    const _Float16* xb = xseq + (size_t)row0 * xrow + (size_t)t * 64;
    v16h xa0 = load_a_frag(xb, (int)xrow, 0, lane);
    v16h xa1 = load_a_frag(xb, (int)xrow, 32, lane);
    v16h h0 = load_a_frag(&hbuf[cur][wv][0], 64, 0, lane);
    v16h h1 = load_a_frag(&hbuf[cur][wv][0], 64, 32, lane);
    if (t + 1 < T_)
      __builtin_prefetch(xseq + (size_t)(row0 + l15) * xrow + (size_t)(t + 1) * 64, 0, 3);
    #pragma unroll
    for (int ct = 0; ct < 4; ct++) {
      v8f acc[4];
      #pragma unroll
      for (int gidx = 0; gidx < 4; gidx++) {
        const int j = gidx * 4 + ct;
        v8f c;
        #pragma unroll
        for (int e = 0; e < 8; e++) c[e] = bs[j];
        c = wmma16(xa0, load_bt_frag(&sWih[(j * 16) * 64], 64, 0, lane), c);
        c = wmma16(xa1, load_bt_frag(&sWih[(j * 16) * 64], 64, 32, lane), c);
        c = wmma16(h0, load_bt_frag(&sWhh[(j * 16) * 64], 64, 0, lane), c);
        c = wmma16(h1, load_bt_frag(&sWhh[(j * 16) * 64], 64, 32, lane), c);
        acc[gidx] = c;
      }
      #pragma unroll
      for (int e = 0; e < 8; e++) {
        float ii = sigf(acc[0][e]), ff = sigf(acc[1][e]);
        float g_ = tanhf_(acc[2][e]), oo = sigf(acc[3][e]);
        float cc = ff * creg[ct][e] + ii * g_;
        creg[ct][e] = cc;
        float hh = oo * tanhf_(cc);
        int row = e + hi * 8, col = ct * 16 + l15;
        hbuf[nxt][wv][row * 64 + col] = (_Float16)hh;
        yseq[((size_t)(row0 + row) * T_ + t) * 64 + col] = (_Float16)hh;
      }
    }
    __syncthreads();
  }
}

// ---------------------------------------------------------------------------
// x_hat = o1(16384x64) @ out_W^T(64x64) + out_b  -> f32 d_out
// ---------------------------------------------------------------------------
__global__ void __launch_bounds__(128)
outgemm_kernel(const _Float16* __restrict__ o1, const _Float16* __restrict__ Wh,
               const float* __restrict__ bias, float* __restrict__ out) {
  const int tid = threadIdx.x, lane = tid & 31, wv = tid >> 5;
  const int hi = (lane >> 4) & 1, l15 = lane & 15;
  const int row0 = (blockIdx.x * 4 + wv) * 16;
  v16h a0 = load_a_frag(o1 + (size_t)row0 * 64, 64, 0, lane);
  v16h a1 = load_a_frag(o1 + (size_t)row0 * 64, 64, 32, lane);
  #pragma unroll
  for (int nt = 0; nt < 4; nt++) {
    float bb = bias[nt * 16 + l15];
    v8f c;
    #pragma unroll
    for (int e = 0; e < 8; e++) c[e] = bb;
    c = wmma16(a0, load_bt_frag(&Wh[(nt * 16) * 64], 64, 0, lane), c);
    c = wmma16(a1, load_bt_frag(&Wh[(nt * 16) * 64], 64, 32, lane), c);
    #pragma unroll
    for (int e = 0; e < 8; e++) {
      int row = row0 + e + hi * 8, col = nt * 16 + l15;
      out[(size_t)row * 64 + col] = c[e];
    }
  }
}

// ---------------------------------------------------------------------------
extern "C" void kernel_launch(void* const* d_in, const int* in_sizes, int n_in,
                              void* d_out, int out_size, void* d_ws,
                              size_t ws_size, hipStream_t stream) {
  (void)in_sizes; (void)n_in; (void)out_size; (void)ws_size;
  const float* x      = (const float*)d_in[0];
  const float* adj1   = (const float*)d_in[1];
  const float* W1     = (const float*)d_in[2];
  const float* b1     = (const float*)d_in[3];
  const float* adj2   = (const float*)d_in[4];
  const float* W2     = (const float*)d_in[5];
  const float* b2     = (const float*)d_in[6];
  const float* ln_g   = (const float*)d_in[7];
  const float* ln_b   = (const float*)d_in[8];
  const float* g0Wih  = (const float*)d_in[9];
  const float* g0Whh  = (const float*)d_in[10];
  const float* g0bih  = (const float*)d_in[11];
  const float* g0bhh  = (const float*)d_in[12];
  const float* g1Wih  = (const float*)d_in[13];
  const float* g1Whh  = (const float*)d_in[14];
  const float* g1bih  = (const float*)d_in[15];
  const float* g1bhh  = (const float*)d_in[16];
  const float* pool_W = (const float*)d_in[17];
  const float* pool_b = (const float*)d_in[18];
  const float* mu_W   = (const float*)d_in[19];
  const float* mu_b   = (const float*)d_in[20];
  const float* logv_W = (const float*)d_in[21];
  const float* logv_b = (const float*)d_in[22];
  const float* dec_W  = (const float*)d_in[23];
  const float* dec_b  = (const float*)d_in[24];
  const float* lWih   = (const float*)d_in[25];
  const float* lWhh   = (const float*)d_in[26];
  const float* lbih   = (const float*)d_in[27];
  const float* lbhh   = (const float*)d_in[28];
  const float* out_W  = (const float*)d_in[29];
  const float* out_b  = (const float*)d_in[30];
  float* out = (float*)d_out;

  char* base = (char*)d_ws;
  size_t off = 0;
  auto alloc = [&](size_t bytes) -> void* {
    off = (off + 255) & ~(size_t)255;
    void* p = base + off;
    off += bytes;
    return p;
  };
  _Float16* A1s    = (_Float16*)alloc(64 * 64 * 2);
  _Float16* A2s    = (_Float16*)alloc(64 * 64 * 2);
  _Float16* W2h    = (_Float16*)alloc(32 * 32 * 2);
  _Float16* outWh  = (_Float16*)alloc(64 * 64 * 2);
  _Float16* g0WihH = (_Float16*)alloc((size_t)2 * 192 * 32 * 2);
  _Float16* g0WhhH = (_Float16*)alloc((size_t)2 * 192 * 64 * 2);
  _Float16* g1WihH = (_Float16*)alloc((size_t)2 * 192 * 128 * 2);
  _Float16* g1WhhH = (_Float16*)alloc((size_t)2 * 192 * 64 * 2);
  _Float16* lWihH  = (_Float16*)alloc((size_t)2 * 256 * 64 * 2);
  _Float16* lWhhH  = (_Float16*)alloc((size_t)2 * 256 * 64 * 2);
  _Float16* h_enc  = (_Float16*)alloc((size_t)4096 * 256 * 32 * 2);   //  64 MB
  _Float16* y0     = (_Float16*)alloc((size_t)4096 * 256 * 128 * 2);  // 256 MB
  _Float16* hlast  = (_Float16*)alloc((size_t)4096 * 128 * 2);
  _Float16* dhbuf  = (_Float16*)alloc(64 * 64 * 2);
  _Float16* o0     = (_Float16*)alloc((size_t)64 * 256 * 64 * 2);
  _Float16* o1     = (_Float16*)alloc((size_t)64 * 256 * 64 * 2);

  // prep
  softmax64_kernel<<<1, 64, 0, stream>>>(adj1, A1s);
  softmax64_kernel<<<1, 64, 0, stream>>>(adj2, A2s);
  auto cvt = [&](const float* s, _Float16* d, int n) {
    cvt_f16_kernel<<<(n + 255) / 256, 256, 0, stream>>>(s, d, n);
  };
  cvt(W2, W2h, 32 * 32);
  cvt(out_W, outWh, 64 * 64);
  cvt(g0Wih, g0WihH, 2 * 192 * 32);
  cvt(g0Whh, g0WhhH, 2 * 192 * 64);
  cvt(g1Wih, g1WihH, 2 * 192 * 128);
  cvt(g1Whh, g1WhhH, 2 * 192 * 64);
  cvt(lWih, lWihH, 2 * 256 * 64);
  cvt(lWhh, lWhhH, 2 * 256 * 64);

  // encoder GNN (B*T items)
  gnn_kernel<<<B_ * T_, 128, 0, stream>>>(x, A1s, A2s, W2h, W1, b1, b2,
                                          ln_g, ln_b, h_enc);

  const size_t gruDyn0 = (size_t)(192 * 32 + 192 * 64) * 2;   // 36 KB
  const size_t gruDyn1 = (size_t)(192 * 128 + 192 * 64) * 2;  // 72 KB
  const size_t lstmDyn = (size_t)(2 * 256 * 64) * 2;          // 64 KB

  // bidirectional GRU layer 0 (full sequence output)
  gru_kernel<32, false><<<64, 128, gruDyn0, stream>>>(
      h_enc, g0WihH, g0WhhH, g0bih, g0bhh, 256, y0, 0, nullptr, 0);
  gru_kernel<32, true><<<64, 128, gruDyn0, stream>>>(
      h_enc, g0WihH + 192 * 32, g0WhhH + 192 * 64, g0bih + 192, g0bhh + 192,
      256, y0, 64, nullptr, 0);
  // GRU layer 1: fwd full scan (final state); bwd = single step on t=T-1
  gru_kernel<128, false><<<64, 128, gruDyn1, stream>>>(
      y0, g1WihH, g1WhhH, g1bih, g1bhh, 256, nullptr, 0, hlast, 0);
  gru_kernel<128, true><<<64, 128, gruDyn1, stream>>>(
      y0, g1WihH + 192 * 128, g1WhhH + 192 * 64, g1bih + 192, g1bhh + 192,
      1, nullptr, 0, hlast, 64);

  float* out_mu = out + (size_t)B_ * T_ * C_;
  float* out_logv = out_mu + B_ * L_;
  pool_kernel<<<1, 256, 0, stream>>>(hlast, pool_W, pool_b, mu_W, mu_b,
                                     logv_W, logv_b, dec_W, dec_b,
                                     out_mu, out_logv, dhbuf);

  // decoder LSTM
  lstm_first_kernel<<<1, 128, lstmDyn, stream>>>(dhbuf, lWihH, lWhhH, lbih,
                                                 lbhh, o0);
  lstm_seq_kernel<<<1, 128, lstmDyn, stream>>>(o0, lWihH + 256 * 64,
                                               lWhhH + 256 * 64, lbih + 256,
                                               lbhh + 256, o1);
  // x_hat
  outgemm_kernel<<<256, 128, 0, stream>>>(o1, outWh, out_b, out);
}
